// SS_MSA_42700564856857
// MI455X (gfx1250) — matrix operand order repaired and checked by reading, not compiled, not run on hardware
//
#include <hip/hip_runtime.h>
#include <hip/hip_bf16.h>
#include <math.h>
#include <stdint.h>

typedef __bf16 bf16_t;
typedef __attribute__((ext_vector_type(16))) __bf16 v16bf;
typedef __attribute__((ext_vector_type(8)))  __bf16 v8bf;
typedef __attribute__((ext_vector_type(4)))  __bf16 v4bf;
typedef __attribute__((ext_vector_type(8)))  float  v8f;

#define NROWS   262144              // 256*32*32 tokens
#define SCALE_F 0.1889822365046136f // 28^-0.5

union U16 { v16bf v; v8bf h[2]; v4bf q[4]; bf16_t e[16]; };
union U4  { v4bf  v; bf16_t e[4]; };

static __device__ __forceinline__ v8f wmma_bf16(v16bf a, v16bf b, v8f c) {
  // D = A(16x32 bf16) * B(32x16 bf16) + C(16x16 f32)
  return __builtin_amdgcn_wmma_f32_16x16x32_bf16(false, a, false, b, (short)0, c, false, false);
}

// A-fragment per-lane pattern: elems 0..7 at p, elems 8..15 at p+16 (8B aligned).
static __device__ __forceinline__ v16bf loadA16(const bf16_t* p) {
  U16 u;
  u.q[0] = *(const v4bf*)(p + 0);
  u.q[1] = *(const v4bf*)(p + 4);
  u.q[2] = *(const v4bf*)(p + 16);
  u.q[3] = *(const v4bf*)(p + 20);
  return u.v;
}
// B-fragment per-lane pattern: elems 0..7 at p, elems 8..15 at p+16 (8B aligned).
static __device__ __forceinline__ v16bf loadB16(const bf16_t* p) {
  U16 u;
  u.q[0] = *(const v4bf*)(p + 0);
  u.q[1] = *(const v4bf*)(p + 4);
  u.q[2] = *(const v4bf*)(p + 16);
  u.q[3] = *(const v4bf*)(p + 20);
  return u.v;
}
// 16 contiguous bf16 at a 16B-aligned pointer (two b128 loads; for LDS).
static __device__ __forceinline__ v16bf loadB16a(const bf16_t* p) {
  U16 u;
  u.h[0] = *(const v8bf*)(p + 0);
  u.h[1] = *(const v8bf*)(p + 8);
  return u.v;
}
static __device__ __forceinline__ v16bf zero_tail4(v16bf x) {
  U16 u; u.v = x;
  u.e[12] = u.e[13] = u.e[14] = u.e[15] = (bf16_t)0.0f;
  return u.v;
}

// ---------------------------------------------------------------------------
// Pre-swizzle Wq|Wkv (224x672) and Wo (224x224) fp32 -> bf16 B-fragment layout.
// fragAll[(kc*42+nt)*32 + lane][16] ; fragWo[(kc*14+nt)*32 + lane][16]
// ---------------------------------------------------------------------------
__global__ __launch_bounds__(256) void k_swizzle(const float* __restrict__ Wq,
                                                 const float* __restrict__ Wkv,
                                                 const float* __restrict__ Wo,
                                                 bf16_t* __restrict__ fragAll,
                                                 bf16_t* __restrict__ fragWo) {
  int t = blockIdx.x * 256 + threadIdx.x;
  if (t < 7 * 42 * 32) {
    int lane = t & 31, tile = t >> 5;
    int nt = tile % 42, kc = tile / 42;
    int col = nt * 16 + (lane & 15);
    int k0  = kc * 32 + ((lane < 16) ? 0 : 8);
    bf16_t* dst = fragAll + (size_t)t * 16;
#pragma unroll
    for (int e = 0; e < 16; ++e) {
      int k = k0 + e + ((e < 8) ? 0 : 8);
      float v = (col < 224) ? Wq[k * 224 + col] : Wkv[k * 448 + (col - 224)];
      dst[e] = (bf16_t)v;
    }
  } else if (t < 7 * 42 * 32 + 7 * 14 * 32) {
    int u = t - 7 * 42 * 32;
    int lane = u & 31, tile = u >> 5;
    int nt = tile % 14, kc = tile / 14;
    int col = nt * 16 + (lane & 15);
    int k0  = kc * 32 + ((lane < 16) ? 0 : 8);
    bf16_t* dst = fragWo + (size_t)u * 16;
#pragma unroll
    for (int e = 0; e < 16; ++e)
      dst[e] = (bf16_t)Wo[(k0 + e + ((e < 8) ? 0 : 8)) * 224 + col];
  }
}

// ---------------------------------------------------------------------------
// QKV projection: q/k/v (bf16, contiguous buffers) written in windowed order
// r' = b*1024 + n*64 + m  with n=(h/8)*4+(w/8), m=(h%8)*8+(w%8)
// B-fragments staged through LDS with async DMA (double-buffered):
// waves 0..6 each own one K-chunk; all 8 waves consume via ds_load_b128.
// ---------------------------------------------------------------------------
__global__ __launch_bounds__(256) void k_qkv(const float* __restrict__ x,
                                             const bf16_t* __restrict__ fragAll,
                                             bf16_t* __restrict__ qkvbase) {
  __shared__ bf16_t Bbuf[2][7][32][16];   // 2 x 7KB staging buffers
  int wave = threadIdx.x >> 5, lane = threadIdx.x & 31;
  int l15 = lane & 15; bool lo = lane < 16;
  int rbase = (blockIdx.x * 8 + wave) * 16;
  // A-fragment row this lane supplies (windowed index -> x spatial row)
  int rp = rbase + l15;
  int b = rp >> 10, idx = rp & 1023, n = idx >> 6, m = idx & 63;
  int hh = (n >> 2) * 8 + (m >> 3), ww = (n & 3) * 8 + (m & 7);
  const float* xr = x + ((size_t)(b << 10) + hh * 32 + ww) * 224;
  int off = lo ? 0 : 8;

  v16bf a[7];
#pragma unroll
  for (int kc = 0; kc < 7; ++kc) {
    const float* p = xr + kc * 32 + off;
    float4 f0 = *(const float4*)(p);
    float4 f1 = *(const float4*)(p + 4);
    float4 f2 = *(const float4*)(p + 16);
    float4 f3 = *(const float4*)(p + 20);
    U16 u;
    u.e[0]  = (bf16_t)f0.x; u.e[1]  = (bf16_t)f0.y; u.e[2]  = (bf16_t)f0.z; u.e[3]  = (bf16_t)f0.w;
    u.e[4]  = (bf16_t)f1.x; u.e[5]  = (bf16_t)f1.y; u.e[6]  = (bf16_t)f1.z; u.e[7]  = (bf16_t)f1.w;
    u.e[8]  = (bf16_t)f2.x; u.e[9]  = (bf16_t)f2.y; u.e[10] = (bf16_t)f2.z; u.e[11] = (bf16_t)f2.w;
    u.e[12] = (bf16_t)f3.x; u.e[13] = (bf16_t)f3.y; u.e[14] = (bf16_t)f3.z; u.e[15] = (bf16_t)f3.w;
    a[kc] = u.v;
  }

  const uint64_t gbase = (uint64_t)(uintptr_t)fragAll;
  bool issuer = (wave < 7);
  // Async DMA of this wave's K-chunk for stage nt into Bbuf[buf] (32B/lane).
  // INST_OFFSET applies to both LDS and global addresses (ISA 08 §4.4).
#define PREFETCH_STAGE(bufi, nti)                                              \
  do {                                                                         \
    if (issuer) {                                                              \
      unsigned ldsa = (unsigned)(uintptr_t)&Bbuf[(bufi)][wave][lane][0];       \
      unsigned gofs = (unsigned)(((wave * 42 + (nti)) * 512 + lane * 16) * 2); \
      asm volatile("global_load_async_to_lds_b128 %0, %1, %2\n\t"              \
                   "global_load_async_to_lds_b128 %0, %1, %2 offset:16"        \
                   :: "v"(ldsa), "v"(gofs), "s"(gbase) : "memory");            \
    }                                                                          \
  } while (0)

  PREFETCH_STAGE(0, 0);

  int rowadd = lo ? 0 : 8;
  const size_t bufE = (size_t)NROWS * 224;
  for (int nt = 0; nt < 42; ++nt) {
    int cur = nt & 1;
    if (nt + 1 < 42) {
      PREFETCH_STAGE(cur ^ 1, nt + 1);
      asm volatile("s_wait_asynccnt 0x2" ::: "memory");  // current stage landed
    } else {
      asm volatile("s_wait_asynccnt 0x0" ::: "memory");
    }
    __syncthreads();                                     // publish Bbuf[cur]

    v16bf bf[7];
#pragma unroll
    for (int kc = 0; kc < 7; ++kc)
      bf[kc] = loadB16a(&Bbuf[cur][kc][lane][0]);
    v8f acc = {};
#pragma unroll
    for (int kc = 0; kc < 7; ++kc)
      acc = wmma_bf16(a[kc], bf[kc], acc);
    __syncthreads();                                     // done reading Bbuf[cur]

    // Branchless q/k/v select: buffers are contiguous in workspace.
    int sel = (nt * 16) / 224;                 // 0:q 1:k 2:v
    int c   = nt * 16 + l15 - sel * 224;
    bf16_t* dbuf = qkvbase + (size_t)sel * bufE;
#pragma unroll
    for (int v = 0; v < 8; ++v)
      dbuf[(size_t)(rbase + rowadd + v) * 224 + c] = (bf16_t)acc[v];
  }
#undef PREFETCH_STAGE
}

// ---------------------------------------------------------------------------
// Branch 1: window attention, seq=64, heads 0..3 over channels 0..111.
// One wave per head (2 heads / block). Output overwrites qbuf channels.
// ---------------------------------------------------------------------------
__global__ __launch_bounds__(64) void k_attn_win(bf16_t* __restrict__ qbuf,
                                                 const bf16_t* __restrict__ kbuf,
                                                 const bf16_t* __restrict__ vbuf,
                                                 const float* __restrict__ pos1) {
  int wv = threadIdx.x >> 5, lane = threadIdx.x & 31;
  int head = blockIdx.y * 2 + wv;
  size_t rbase = (size_t)blockIdx.x * 64;
  int hch = head * 28;
  __shared__ float  S[2][64][64];   // rolled sim
  __shared__ bf16_t P[2][64][64];   // softmaxed probs (bf16)
  __shared__ bf16_t Vt[2][32][64];  // V transposed, d rows 28..31 zero
  bool lo = lane < 16; int l15 = lane & 15;

  // Build Vt: lane owns columns j = lane, lane+32 (7 vector loads per column)
#pragma unroll
  for (int rr = 0; rr < 2; ++rr) {
    int j = lane + rr * 32;
    const bf16_t* vr = vbuf + (rbase + j) * 224 + hch;
    U4 c[7];
#pragma unroll
    for (int t = 0; t < 7; ++t) c[t].v = *(const v4bf*)(vr + t * 4);
#pragma unroll
    for (int t = 0; t < 7; ++t)
#pragma unroll
      for (int e = 0; e < 4; ++e) Vt[wv][t * 4 + e][j] = c[t].e[e];
#pragma unroll
    for (int d = 28; d < 32; ++d) Vt[wv][d][j] = (bf16_t)0.0f;
  }

  // Q fragments (4 M-tiles, K=d padded 28->32)
  v16bf qa[4];
#pragma unroll
  for (int mt = 0; mt < 4; ++mt) {
    const bf16_t* qr = qbuf + (rbase + mt * 16 + l15) * 224 + hch + (lo ? 0 : 8);
    v16bf t = loadA16(qr);
    if (!lo) t = zero_tail4(t);   // d 28..31 -> 0
    qa[mt] = t;
  }

  // S = roll(Q K^T * scale + pos, (1,1))  (roll applied at scatter)
  for (int ntj = 0; ntj < 4; ++ntj) {
    const bf16_t* kr = kbuf + (rbase + ntj * 16 + l15) * 224 + hch + (lo ? 0 : 8);
    v16bf kf = loadB16(kr);
    if (!lo) kf = zero_tail4(kf);
    int j = ntj * 16 + l15;
#pragma unroll
    for (int mt = 0; mt < 4; ++mt) {
      v8f acc = {};
      acc = wmma_bf16(qa[mt], kf, acc);
      int ibase = mt * 16 + (lo ? 0 : 8);
#pragma unroll
      for (int v = 0; v < 8; ++v) {
        int i = ibase + v;
        S[wv][(i + 1) & 63][(j + 1) & 63] =
            acc[v] * SCALE_F + pos1[head * 4096 + i * 64 + j];
      }
    }
  }
  __syncthreads();

  // Row softmax (roll already applied)
#pragma unroll
  for (int rr = 0; rr < 2; ++rr) {
    int r = lane + rr * 32;
    float* row = &S[wv][r][0];
    float mx = -3.0e38f;
    for (int j = 0; j < 64; j += 4) {
      float4 f = *(const float4*)(row + j);
      mx = fmaxf(mx, fmaxf(fmaxf(f.x, f.y), fmaxf(f.z, f.w)));
    }
    float sum = 0.f;
    for (int j = 0; j < 64; j += 4) {
      float4 f = *(float4*)(row + j);
      f.x = __expf(f.x - mx); f.y = __expf(f.y - mx);
      f.z = __expf(f.z - mx); f.w = __expf(f.w - mx);
      sum += f.x + f.y + f.z + f.w;
      *(float4*)(row + j) = f;
    }
    float inv = 1.0f / sum;
    for (int j = 0; j < 64; j += 4) {
      float4 f = *(const float4*)(row + j);
      U4 w;
      w.e[0] = (bf16_t)(f.x * inv); w.e[1] = (bf16_t)(f.y * inv);
      w.e[2] = (bf16_t)(f.z * inv); w.e[3] = (bf16_t)(f.w * inv);
      *(v4bf*)(&P[wv][r][j]) = w.v;
    }
  }
  __syncthreads();

  // O = P @ V  (Kdim = 64 -> 2 chunks; N = d padded 28->32)
  for (int nt2 = 0; nt2 < 2; ++nt2) {
    int d = nt2 * 16 + l15;
#pragma unroll
    for (int mt = 0; mt < 4; ++mt) {
      v16bf pf0 = loadA16(&P[wv][mt * 16 + l15][0 + (lo ? 0 : 8)]);
      v16bf pf1 = loadA16(&P[wv][mt * 16 + l15][32 + (lo ? 0 : 8)]);
      v16bf vf0 = loadB16(&Vt[wv][nt2 * 16 + l15][0 + (lo ? 0 : 8)]);
      v16bf vf1 = loadB16(&Vt[wv][nt2 * 16 + l15][32 + (lo ? 0 : 8)]);
      v8f acc = {};
      acc = wmma_bf16(pf0, vf0, acc);
      acc = wmma_bf16(pf1, vf1, acc);
      if (d < 28) {
#pragma unroll
        for (int v = 0; v < 8; ++v)
          qbuf[(rbase + mt * 16 + (lo ? 0 : 8) + v) * 224 + hch + d] = (bf16_t)acc[v];
      }
    }
  }
}

// ---------------------------------------------------------------------------
// Branch 2: grid attention, seq=16 (windows), heads over channels 112..223.
// One wave per head (4 heads / block). Group = (b, m); rows stride 64.
// ---------------------------------------------------------------------------
__global__ __launch_bounds__(128) void k_attn_grid(bf16_t* __restrict__ qbuf,
                                                   const bf16_t* __restrict__ kbuf,
                                                   const bf16_t* __restrict__ vbuf,
                                                   const float* __restrict__ pos2) {
  int head = threadIdx.x >> 5, lane = threadIdx.x & 31;
  int g = blockIdx.x; int b = g >> 6, m = g & 63;
  size_t rb = (size_t)b * 1024 + m;     // + n*64 per window
  int hch = 112 + head * 28;
  __shared__ float  S[4][16][16];
  __shared__ bf16_t P[4][16][32];       // cols 16..31 zero (K pad)
  __shared__ bf16_t Vt[4][32][32];      // rows 28..31 & cols 16..31 zero
  bool lo = lane < 16; int l15 = lane & 15;

  if (lo) {
    int j = l15;
    const bf16_t* vr = vbuf + (rb + (size_t)j * 64) * 224 + hch;
    U4 c[7];
#pragma unroll
    for (int t = 0; t < 7; ++t) c[t].v = *(const v4bf*)(vr + t * 4);
#pragma unroll
    for (int t = 0; t < 7; ++t)
#pragma unroll
      for (int e = 0; e < 4; ++e) Vt[head][t * 4 + e][j] = c[t].e[e];
#pragma unroll
    for (int d = 28; d < 32; ++d) Vt[head][d][j] = (bf16_t)0.0f;
  } else {
    int j = 16 + l15;
#pragma unroll
    for (int d = 0; d < 32; ++d) Vt[head][d][j] = (bf16_t)0.0f;
  }

  const bf16_t* qp = qbuf + (rb + (size_t)l15 * 64) * 224 + hch + (lo ? 0 : 8);
  v16bf qa = loadA16(qp);
  const bf16_t* kp = kbuf + (rb + (size_t)l15 * 64) * 224 + hch + (lo ? 0 : 8);
  v16bf kf = loadB16(kp);
  if (!lo) { qa = zero_tail4(qa); kf = zero_tail4(kf); }

  v8f acc = {};
  acc = wmma_bf16(qa, kf, acc);
  int j = l15, ib = lo ? 0 : 8;
#pragma unroll
  for (int v = 0; v < 8; ++v) {
    int i = ib + v;
    S[head][(i + 1) & 15][(j + 1) & 15] =
        acc[v] * SCALE_F + pos2[head * 256 + i * 16 + j];
  }
  __syncthreads();

  if (lo) {
    int r = l15;
    float* row = &S[head][r][0];
    float4 f0 = *(const float4*)(row + 0);
    float4 f1 = *(const float4*)(row + 4);
    float4 f2 = *(const float4*)(row + 8);
    float4 f3 = *(const float4*)(row + 12);
    float mx = fmaxf(fmaxf(fmaxf(f0.x, f0.y), fmaxf(f0.z, f0.w)),
               fmaxf(fmaxf(fmaxf(f1.x, f1.y), fmaxf(f1.z, f1.w)),
               fmaxf(fmaxf(fmaxf(f2.x, f2.y), fmaxf(f2.z, f2.w)),
                     fmaxf(fmaxf(f3.x, f3.y), fmaxf(f3.z, f3.w)))));
    f0.x = __expf(f0.x - mx); f0.y = __expf(f0.y - mx); f0.z = __expf(f0.z - mx); f0.w = __expf(f0.w - mx);
    f1.x = __expf(f1.x - mx); f1.y = __expf(f1.y - mx); f1.z = __expf(f1.z - mx); f1.w = __expf(f1.w - mx);
    f2.x = __expf(f2.x - mx); f2.y = __expf(f2.y - mx); f2.z = __expf(f2.z - mx); f2.w = __expf(f2.w - mx);
    f3.x = __expf(f3.x - mx); f3.y = __expf(f3.y - mx); f3.z = __expf(f3.z - mx); f3.w = __expf(f3.w - mx);
    float sum = f0.x + f0.y + f0.z + f0.w + f1.x + f1.y + f1.z + f1.w +
                f2.x + f2.y + f2.z + f2.w + f3.x + f3.y + f3.z + f3.w;
    float inv = 1.0f / sum;
    U4 w;
    w.e[0] = (bf16_t)(f0.x * inv); w.e[1] = (bf16_t)(f0.y * inv);
    w.e[2] = (bf16_t)(f0.z * inv); w.e[3] = (bf16_t)(f0.w * inv);
    *(v4bf*)(&P[head][r][0]) = w.v;
    w.e[0] = (bf16_t)(f1.x * inv); w.e[1] = (bf16_t)(f1.y * inv);
    w.e[2] = (bf16_t)(f1.z * inv); w.e[3] = (bf16_t)(f1.w * inv);
    *(v4bf*)(&P[head][r][4]) = w.v;
    w.e[0] = (bf16_t)(f2.x * inv); w.e[1] = (bf16_t)(f2.y * inv);
    w.e[2] = (bf16_t)(f2.z * inv); w.e[3] = (bf16_t)(f2.w * inv);
    *(v4bf*)(&P[head][r][8]) = w.v;
    w.e[0] = (bf16_t)(f3.x * inv); w.e[1] = (bf16_t)(f3.y * inv);
    w.e[2] = (bf16_t)(f3.z * inv); w.e[3] = (bf16_t)(f3.w * inv);
    *(v4bf*)(&P[head][r][12]) = w.v;
  } else {
    int r = l15;
    U4 w; w.e[0] = w.e[1] = w.e[2] = w.e[3] = (bf16_t)0.0f;
#pragma unroll
    for (int jj = 16; jj < 32; jj += 4) *(v4bf*)(&P[head][r][jj]) = w.v;
  }
  __syncthreads();

  v16bf pf = loadA16(&P[head][l15][lo ? 0 : 8]);
#pragma unroll
  for (int nt2 = 0; nt2 < 2; ++nt2) {
    v16bf vf = loadB16(&Vt[head][nt2 * 16 + l15][lo ? 0 : 8]);
    v8f o = {};
    o = wmma_bf16(pf, vf, o);
    int d = nt2 * 16 + l15;
    if (d < 28) {
#pragma unroll
      for (int v = 0; v < 8; ++v)
        qbuf[(rb + (size_t)(ib + v) * 64) * 224 + hch + d] = (bf16_t)o[v];
    }
  }
}

// ---------------------------------------------------------------------------
// Output projection + bias + window merge -> d_out (fp32)
// ---------------------------------------------------------------------------
__global__ __launch_bounds__(256) void k_oproj(const bf16_t* __restrict__ obuf,
                                               const bf16_t* __restrict__ fragWo,
                                               const float* __restrict__ bo,
                                               float* __restrict__ out) {
  int wave = threadIdx.x >> 5, lane = threadIdx.x & 31;
  int l15 = lane & 15; bool lo = lane < 16;
  int rbase = (blockIdx.x * 8 + wave) * 16;
  const bf16_t* ar = obuf + (size_t)(rbase + l15) * 224 + (lo ? 0 : 8);
  v16bf a[7];
#pragma unroll
  for (int kc = 0; kc < 7; ++kc) a[kc] = loadA16(ar + kc * 32);

  size_t dstRow[8];
#pragma unroll
  for (int v = 0; v < 8; ++v) {
    int rp = rbase + (lo ? 0 : 8) + v;
    int b = rp >> 10, idx = rp & 1023, n = idx >> 6, m = idx & 63;
    int hh = (n >> 2) * 8 + (m >> 3), ww = (n & 3) * 8 + (m & 7);
    dstRow[v] = ((size_t)(b << 10) + hh * 32 + ww) * 224;
  }
  for (int nt = 0; nt < 14; ++nt) {
    v16bf bf[7];
#pragma unroll
    for (int kc = 0; kc < 7; ++kc)
      bf[kc] = loadB16(fragWo + ((size_t)(kc * 14 + nt) * 32 + lane) * 16);
    v8f acc = {};
#pragma unroll
    for (int kc = 0; kc < 7; ++kc)
      acc = wmma_bf16(a[kc], bf[kc], acc);
    int c = nt * 16 + l15;
    float bias = bo[c];
#pragma unroll
    for (int v = 0; v < 8; ++v)
      out[dstRow[v] + c] = acc[v] + bias;
  }
}

extern "C" void kernel_launch(void* const* d_in, const int* in_sizes, int n_in,
                              void* d_out, int out_size, void* d_ws, size_t ws_size,
                              hipStream_t stream) {
  (void)in_sizes; (void)n_in; (void)out_size; (void)ws_size;
  const float* x   = (const float*)d_in[0];
  const float* Wq  = (const float*)d_in[1];
  const float* Wkv = (const float*)d_in[2];
  const float* Wo  = (const float*)d_in[3];
  const float* bo  = (const float*)d_in[4];
  const float* p1  = (const float*)d_in[5];
  const float* p2  = (const float*)d_in[6];
  float* out = (float*)d_out;

  const size_t bufE = (size_t)NROWS * 224;       // bf16 elements per buffer
  bf16_t* qbuf    = (bf16_t*)d_ws;               // q | k | v contiguous
  bf16_t* kbuf    = qbuf + bufE;
  bf16_t* vbuf    = kbuf + bufE;
  bf16_t* fragAll = vbuf + bufE;                 // 7*42*32*16 bf16
  bf16_t* fragWo  = fragAll + (size_t)7 * 42 * 32 * 16;

  k_swizzle  <<<49, 256, 0, stream>>>(Wq, Wkv, Wo, fragAll, fragWo);
  k_qkv      <<<NROWS / 128, 256, 0, stream>>>(x, fragAll, qbuf);
  k_attn_win <<<dim3(4096, 2), 64, 0, stream>>>(qbuf, kbuf, vbuf, p1);
  k_attn_grid<<<16384, 128, 0, stream>>>(qbuf, kbuf, vbuf, p2);
  k_oproj    <<<NROWS / 128, 256, 0, stream>>>(qbuf, fragWo, bo, out);
}
